// ChebEncoder_37726992728724
// MI455X (gfx1250) — matrix operand order repaired and checked by reading, not compiled, not run on hardware
//
#include <hip/hip_runtime.h>
#include <hip/hip_bf16.h>

typedef __attribute__((ext_vector_type(16))) _Float16 v16h;
typedef __attribute__((ext_vector_type(8)))  _Float16 v8h;
typedef __attribute__((ext_vector_type(4)))  _Float16 v4h;
typedef __attribute__((ext_vector_type(8)))  float    v8f;

#define H          256          // embed == hidden == 256
#define LDS_STRIDE 264          // 256 + 8 halves pad: row stride 528B -> bank-conflict-free
#define M_BLOCK    80           // 100000 = 1250 * 80, no tail
#define N_KT       8            // 256 / 32  (K tiles per GEMM)
#define N_NT       16           // 256 / 16  (N tiles)
#define N_MS       5            // 80 / 16   (M tiles per block)

// ---------------------------------------------------------------------------
// Weight prepack: fp32 row-major W[k][n] (256x256) -> WMMA B-fragment layout.
// Fragment(kt,nt) = 32 lanes x 16 halves, contiguous per lane (32B/lane):
//   lane<16 : N = nt*16+lane,      K = kt*32 + h        (h = 0..15)
//   lane>=16: N = nt*16+(lane-16), K = kt*32 + 16 + h
// (this code's contiguous K order, as in load_a_frag; one K-row pair packed lo/hi per VGPR)
// ---------------------------------------------------------------------------
__global__ void prepack_w(const float* __restrict__ W, _Float16* __restrict__ dst) {
    int g    = blockIdx.x * 256 + threadIdx.x;   // 0 .. 65535
    int h    =  g        & 15;
    int lane = (g >> 4)  & 31;
    int nt   = (g >> 9)  & 15;
    int kt   =  g >> 13;
    int k = kt * 32 + ((lane & 16) ? 16 : 0) + h;
    int n = nt * 16 + (lane & 15);
    dst[g] = (_Float16)W[k * H + n];
}

// Gather one 16x32 f16 A fragment from an LDS tile (row-major, LDS_STRIDE).
// This code's K order: lane<16 row=lane,    K = {0..15}
//                      lane>=16 row=lane-16, K = {16..31}
// -> two aligned 16B ds_load_b128 chunks per lane.
__device__ __forceinline__ v16h load_a_frag(const _Float16* lds, int row_base,
                                            int kt, int lane) {
    int off = (row_base + (lane & 15)) * LDS_STRIDE + kt * 32 + ((lane & 16) ? 16 : 0);
    v8h c0 = *(const v8h*)(lds + off);
    v8h c1 = *(const v8h*)(lds + off + 8);
    return __builtin_shufflevector(c0, c1, 0,1,2,3,4,5,6,7,8,9,10,11,12,13,14,15);
}

// ---------------------------------------------------------------------------
// Fused: out = relu(emb@W1 + b1) @ W2 + b2, per-block tile of 80 rows.
// 8 waves; wave w owns n-tiles {2w, 2w+1} across all 5 m-tiles.
// ---------------------------------------------------------------------------
__global__ __launch_bounds__(256)
void cheb_fused(const float* __restrict__ emb,
                const float* __restrict__ b1,
                const float* __restrict__ b2,
                const _Float16* __restrict__ Wpk1,
                const _Float16* __restrict__ Wpk2,
                float* __restrict__ out) {
    __shared__ _Float16 ldsA[M_BLOCK * LDS_STRIDE];   // emb tile (f16)
    __shared__ _Float16 ldsB[M_BLOCK * LDS_STRIDE];   // layer-1 output tile (f16)

    const int tid  = threadIdx.x;
    const int lane = tid & 31;
    const int wave = tid >> 5;
    const int m0   = blockIdx.x * M_BLOCK;
    const int nt0  = wave * 2;               // first of this wave's two n-tiles
    const int ncol = lane & 15;              // C/D fragment column within n-tile
    const int rhi  = (lane & 16) ? 8 : 0;    // C/D fragment row offset for hi lanes

    // ---- stage emb tile: 80x256 fp32 -> f16 LDS (coalesced float4 loads) ----
    #pragma unroll
    for (int it = 0; it < (M_BLOCK * H) / (256 * 4); ++it) {   // 20 iters
        int idx = it * 1024 + tid * 4;
        int row = idx >> 8;
        int col = idx & 255;
        const float4 v = *(const float4*)(emb + (size_t)(m0 + row) * H + col);
        v4h hv;
        hv[0] = (_Float16)v.x; hv[1] = (_Float16)v.y;
        hv[2] = (_Float16)v.z; hv[3] = (_Float16)v.w;
        *(v4h*)(ldsA + row * LDS_STRIDE + col) = hv;
    }
    __syncthreads();

    // =========================== layer 1 ===========================
    {
        float bs0 = b1[(nt0 + 0) * 16 + ncol];
        float bs1 = b1[(nt0 + 1) * 16 + ncol];
        v8f acc[N_MS][2];
        #pragma unroll
        for (int ms = 0; ms < N_MS; ++ms) {
            #pragma unroll
            for (int r = 0; r < 8; ++r) { acc[ms][0][r] = bs0; acc[ms][1][r] = bs1; }
        }
        #pragma unroll
        for (int kt = 0; kt < N_KT; ++kt) {
            v16h bf0 = *(const v16h*)(Wpk1 + ((kt * N_NT + nt0 + 0) * 32 + lane) * 16);
            v16h bf1 = *(const v16h*)(Wpk1 + ((kt * N_NT + nt0 + 1) * 32 + lane) * 16);
            #pragma unroll
            for (int ms = 0; ms < N_MS; ++ms) {
                v16h a = load_a_frag(ldsA, ms * 16, kt, lane);
                acc[ms][0] = __builtin_amdgcn_wmma_f32_16x16x32_f16(
                    false, a, false, bf0, (short)0, acc[ms][0], false, false);
                acc[ms][1] = __builtin_amdgcn_wmma_f32_16x16x32_f16(
                    false, a, false, bf1, (short)0, acc[ms][1], false, false);
            }
        }
        // ReLU + spill x tile to LDS as f16 (C/D layout: M = r + rhi, N = nt*16+ncol)
        #pragma unroll
        for (int ms = 0; ms < N_MS; ++ms) {
            #pragma unroll
            for (int j = 0; j < 2; ++j) {
                int n = (nt0 + j) * 16 + ncol;
                #pragma unroll
                for (int r = 0; r < 8; ++r) {
                    float v = acc[ms][j][r];
                    v = v > 0.0f ? v : 0.0f;
                    ldsB[(ms * 16 + rhi + r) * LDS_STRIDE + n] = (_Float16)v;
                }
            }
        }
    }
    __syncthreads();

    // =========================== layer 2 ===========================
    {
        float bs0 = b2[(nt0 + 0) * 16 + ncol];
        float bs1 = b2[(nt0 + 1) * 16 + ncol];
        v8f acc[N_MS][2];
        #pragma unroll
        for (int ms = 0; ms < N_MS; ++ms) {
            #pragma unroll
            for (int r = 0; r < 8; ++r) { acc[ms][0][r] = bs0; acc[ms][1][r] = bs1; }
        }
        #pragma unroll
        for (int kt = 0; kt < N_KT; ++kt) {
            v16h bf0 = *(const v16h*)(Wpk2 + ((kt * N_NT + nt0 + 0) * 32 + lane) * 16);
            v16h bf1 = *(const v16h*)(Wpk2 + ((kt * N_NT + nt0 + 1) * 32 + lane) * 16);
            #pragma unroll
            for (int ms = 0; ms < N_MS; ++ms) {
                v16h a = load_a_frag(ldsB, ms * 16, kt, lane);
                acc[ms][0] = __builtin_amdgcn_wmma_f32_16x16x32_f16(
                    false, a, false, bf0, (short)0, acc[ms][0], false, false);
                acc[ms][1] = __builtin_amdgcn_wmma_f32_16x16x32_f16(
                    false, a, false, bf1, (short)0, acc[ms][1], false, false);
            }
        }
        // fp32 stores: per VGPR r, each 16-lane half writes 16 contiguous floats
        #pragma unroll
        for (int ms = 0; ms < N_MS; ++ms) {
            #pragma unroll
            for (int j = 0; j < 2; ++j) {
                int n = (nt0 + j) * 16 + ncol;
                #pragma unroll
                for (int r = 0; r < 8; ++r) {
                    out[(size_t)(m0 + ms * 16 + rhi + r) * H + n] = acc[ms][j][r];
                }
            }
        }
    }
}

extern "C" void kernel_launch(void* const* d_in, const int* in_sizes, int n_in,
                              void* d_out, int out_size, void* d_ws, size_t ws_size,
                              hipStream_t stream) {
    const float* emb = (const float*)d_in[0];   // [100000,256] f32
    const float* W1  = (const float*)d_in[1];   // [256,256]    f32
    const float* b1  = (const float*)d_in[2];   // [256]        f32
    const float* W2  = (const float*)d_in[3];   // [256,256]    f32
    const float* b2  = (const float*)d_in[4];   // [256]        f32
    // d_in[5] = prop_edge_index: unused at K=1 (no Laplacian term in reference)

    _Float16* Wpk1 = (_Float16*)d_ws;           // 65536 halves (128 KB)
    _Float16* Wpk2 = Wpk1 + 65536;              // 65536 halves (128 KB)

    prepack_w<<<256, 256, 0, stream>>>(W1, Wpk1);
    prepack_w<<<256, 256, 0, stream>>>(W2, Wpk2);
    cheb_fused<<<1250, 256, 0, stream>>>(emb, b1, b2, Wpk1, Wpk2, (float*)d_out);
}
